// ConditionalRandomField_84387517432160
// MI455X (gfx1250) — matrix operand order repaired and checked by reading, not compiled, not run on hardware
//
#include <hip/hip_runtime.h>

// ---------------------------------------------------------------------------
// CRF NLL on MI455X (gfx1250, wave32).
//
// log-semiring recurrence -> shifted-exp + REAL matmul on the WMMA pipe:
//   alpha'[b,j] = m_b + log( sum_i exp(alpha[b,i]-m_b) * exp(trans[i,j]) ) + emit[b,j]
// exp(trans) lives in registers as v_wmma_f32_16x16x32_f16 B-fragments
// (8 WMMA / step / wave, f32 accumulate); transcendental count is cut 64x vs
// the naive logsumexp. One wave owns a 16-batch tile; 64 blocks cover B=1024.
// alpha is LDS-resident (stride-68 pad: conflict-free banks, 16B aligned).
// Round-2 refinements: branch-free masked write-back (v_cndmask), b128 LDS
// loads for the A-source row, raw v_exp_f32 / v_log_f32 via amdgcn builtins.
// ---------------------------------------------------------------------------

typedef __attribute__((ext_vector_type(16))) _Float16 v16h;
typedef __attribute__((ext_vector_type(8)))  float    v8f;

#define Bn 1024
#define Ls 512
#define Tn 64
#define PAD 68           // 68 % 64 == 4 distinct banks per row; 272 B row pitch (16B mult)
#define LOG2E 1.4426950408889634f
#define LN2   0.6931471805599453f

// ---------------- gold path score: one wave per batch --------------------
__global__ __launch_bounds__(32) void crf_gold_kernel(
    const float* __restrict__ logits, const int* __restrict__ tags,
    const int* __restrict__ mask, const float* __restrict__ trans,
    float* __restrict__ gold)
{
  const int b    = blockIdx.x;
  const int lane = threadIdx.x;
  const int*   tg = tags + (size_t)b * Ls;
  const int*   mk = mask + (size_t)b * Ls;
  const float* lg = logits + (size_t)b * Ls * Tn;

  float acc = 0.0f;
  for (int l = lane; l < Ls; l += 32) {
    const float m = (float)mk[l];
    const int   t = tg[l];
    acc += lg[(size_t)l * Tn + t] * m;                 // emission at gold tag
    if (l >= 1) acc += trans[tg[l - 1] * Tn + t] * m;  // gold transition
  }
  for (int o = 16; o >= 1; o >>= 1) acc += __shfl_xor(acc, o, 32);
  if (lane == 0) gold[b] = acc;
}

// ---------------- forward recurrence: one wave per 16 batches ------------
__global__ __launch_bounds__(32) void crf_forward_kernel(
    const float* __restrict__ logits, const int* __restrict__ mask,
    const float* __restrict__ trans, const float* __restrict__ gold,
    float* __restrict__ out)
{
  const int lane = threadIdx.x;
  const int base = blockIdx.x * 16;   // 64 blocks x 16 batch rows
  const int r    = lane & 15;         // batch row this lane owns for A / rowmax
  const int h    = lane >> 4;         // wave half
  const int ab   = h * 8;             // A-layout / D-layout row offset

  __shared__ __align__(16) float alpha[16][PAD];
  __shared__ __align__(16) float emitS[16][PAD];
  __shared__ __align__(16) float rowm[16];
  __shared__ __align__(16) int   maskS[16];

  // --- B fragments: expT = exp(trans) in WMMA-B f16 layout, register-resident.
  // Dense B 32x16 f16: lanes 0-15 hold K=0..7,16..23, lanes 16-31 hold K=8..15,24..31; N = lane&15.
  v16h bfrag[4][2];
  {
    const int kb = h * 8;
#pragma unroll
    for (int jt = 0; jt < 4; ++jt)
#pragma unroll
      for (int kc = 0; kc < 2; ++kc) {
        v16h t;
#pragma unroll
        for (int e = 0; e < 16; ++e) {
          const int K = kc * 32 + (e >> 3) * 16 + kb + (e & 7);          // reduction index i
          const int N = jt * 16 + r;               // output tag j
          t[e] = (_Float16)__builtin_amdgcn_exp2f(trans[K * Tn + N] * LOG2E);
        }
        bfrag[jt][kc] = t;
      }
  }

  // --- alpha[:, :] = logits[:, l=0, :]
#pragma unroll
  for (int it = 0; it < 8; ++it) {
    const int idx4 = it * 32 + lane;               // float4 index within 16x64 tile
    const int row  = idx4 >> 4;
    const int col  = (idx4 & 15) * 4;
    const float4 v = *(const float4*)&logits[((size_t)(base + row) * Ls + 0) * Tn + col];
    *(float4*)&alpha[row][col] = v;
  }
  __syncthreads();

  // --- sequential scan over timesteps
  for (int l = 1; l < Ls; ++l) {
    // stage emission tile (coalesced b128) + per-row mask
#pragma unroll
    for (int it = 0; it < 8; ++it) {
      const int idx4 = it * 32 + lane;
      const int row  = idx4 >> 4;
      const int col  = (idx4 & 15) * 4;
      const float4 v = *(const float4*)&logits[((size_t)(base + row) * Ls + l) * Tn + col];
      *(float4*)&emitS[row][col] = v;
    }
    if (lane < 16) maskS[lane] = mask[(size_t)(base + lane) * Ls + l];

    // prefetch next timestep's emissions (global_prefetch_b8)
    if (l + 1 < Ls)
      __builtin_prefetch(&logits[((size_t)(base + r) * Ls + (l + 1)) * Tn + h * 32], 0, 0);

    // A-source values: 4 contiguous runs of 8 floats -> 8 x ds_load_b128.
    // Half-wave subsets union to the full row, so shfl_xor gives exact rowmax.
    float av[32];
#pragma unroll
    for (int g = 0; g < 4; ++g) {
      const int K0 = g * 16 + ab;
      const float4 x = *(const float4*)&alpha[r][K0];
      const float4 y = *(const float4*)&alpha[r][K0 + 4];
      av[g*8+0]=x.x; av[g*8+1]=x.y; av[g*8+2]=x.z; av[g*8+3]=x.w;
      av[g*8+4]=y.x; av[g*8+5]=y.y; av[g*8+6]=y.z; av[g*8+7]=y.w;
    }
    float pm = av[0];
#pragma unroll
    for (int k = 1; k < 32; ++k) pm = fmaxf(pm, av[k]);
    pm = fmaxf(pm, __shfl_xor(pm, 16, 32));
    if (lane < 16) rowm[lane] = pm;

    // A fragments: p = exp(alpha - rowmax) in f16 WMMA-A layout
    // (16-bit A 16x32: lane<16 -> K = e / 16+e-8 ; lane>=16 -> +8)
    v16h afrag[2];
#pragma unroll
    for (int kc = 0; kc < 2; ++kc) {
      v16h t;
#pragma unroll
      for (int e = 0; e < 16; ++e) {
        const int gi = kc * 2 + (e >> 3);
        t[e] = (_Float16)__builtin_amdgcn_exp2f((av[gi * 8 + (e & 7)] - pm) * LOG2E);
      }
      afrag[kc] = t;
    }
    __syncthreads();  // emitS/maskS/rowm visible; all cross-lane alpha reads done

    // per-row shift & mask into registers (vector LDS loads, no per-elem branch)
    float rm[8]; int mkv[8];
    {
      const float4 a0 = *(const float4*)&rowm[ab];
      const float4 a1 = *(const float4*)&rowm[ab + 4];
      rm[0]=a0.x; rm[1]=a0.y; rm[2]=a0.z; rm[3]=a0.w;
      rm[4]=a1.x; rm[5]=a1.y; rm[6]=a1.z; rm[7]=a1.w;
      const int4 m0 = *(const int4*)&maskS[ab];
      const int4 m1 = *(const int4*)&maskS[ab + 4];
      mkv[0]=m0.x; mkv[1]=m0.y; mkv[2]=m0.z; mkv[3]=m0.w;
      mkv[4]=m1.x; mkv[5]=m1.y; mkv[6]=m1.z; mkv[7]=m1.w;
    }

    // S[b,j] = sum_i p[b,i] * expT[i,j] on the WMMA pipe, f32 accumulate.
    // Write-back is branch-free: each (row,col) below is touched only by this
    // lane this step, so read-old -> cndmask -> store is race-free.
#pragma unroll
    for (int jt = 0; jt < 4; ++jt) {
      v8f acc = {};
      acc = __builtin_amdgcn_wmma_f32_16x16x32_f16(false, afrag[0], false, bfrag[jt][0],
                                                   (short)0, acc, false, false);
      acc = __builtin_amdgcn_wmma_f32_16x16x32_f16(false, afrag[1], false, bfrag[jt][1],
                                                   (short)0, acc, false, false);
      const int j = jt * 16 + r;                  // D layout: N = lane&15
#pragma unroll
      for (int v = 0; v < 8; ++v) {
        const int   r2 = v + ab;                  // D layout: M = v + 8*half
        const float o  = alpha[r2][j];
        const float nv = rm[v] + __builtin_amdgcn_logf(acc[v]) * LN2 + emitS[r2][j];
        alpha[r2][j] = (mkv[v] > 0) ? nv : o;     // jnp.where(mask>0, new, old)
      }
    }
    __syncthreads();  // alpha writes visible before next step's reads
  }

  // --- logZ[b] = logsumexp_j alpha[b,j]; out = logZ - gold
  float fv[32];
#pragma unroll
  for (int q = 0; q < 8; ++q) {
    const float4 x = *(const float4*)&alpha[r][h * 32 + q * 4];
    fv[q*4+0]=x.x; fv[q*4+1]=x.y; fv[q*4+2]=x.z; fv[q*4+3]=x.w;
  }
  float fm = fv[0];
#pragma unroll
  for (int k = 1; k < 32; ++k) fm = fmaxf(fm, fv[k]);
  fm = fmaxf(fm, __shfl_xor(fm, 16, 32));
  float s = 0.0f;
#pragma unroll
  for (int k = 0; k < 32; ++k) s += __builtin_amdgcn_exp2f((fv[k] - fm) * LOG2E);
  s += __shfl_xor(s, 16, 32);
  if (lane < 16) {
    const int b = base + r;
    out[b] = fm + __builtin_amdgcn_logf(s) * LN2 - gold[b];
  }
}

extern "C" void kernel_launch(void* const* d_in, const int* in_sizes, int n_in,
                              void* d_out, int out_size, void* d_ws, size_t ws_size,
                              hipStream_t stream) {
  const float* logits = (const float*)d_in[0];  // [B, L, T] f32
  const int*   tags   = (const int*)d_in[1];    // [B, L] int
  const int*   mask   = (const int*)d_in[2];    // [B, L] int32
  const float* trans  = (const float*)d_in[3];  // [T, T] f32
  float* out  = (float*)d_out;                  // [B] f32
  float* gold = (float*)d_ws;                   // [B] f32 scratch

  crf_gold_kernel<<<Bn, 32, 0, stream>>>(logits, tags, mask, trans, gold);
  crf_forward_kernel<<<Bn / 16, 32, 0, stream>>>(logits, mask, trans, gold, out);
  (void)in_sizes; (void)n_in; (void)out_size; (void)ws_size;
}
